// OptionIIModel_78151224918177
// MI455X (gfx1250) — matrix operand order repaired and checked by reading, not compiled, not run on hardware
//
#include <hip/hip_runtime.h>

typedef _Float16 half_t;
typedef __attribute__((ext_vector_type(16))) _Float16 v16h;
typedef __attribute__((ext_vector_type(8)))  _Float16 v8h;
typedef __attribute__((ext_vector_type(8)))  float    v8f;

#define NVOX   (48*48*48)
#define NWAVES 8

// Output block offsets (floats), in reference return order:
// atom_flux[B,6,8,v], material_momentum[B,6,8,3,v], material_ke[B,6,8,v],
// force_momentum[B,6,3,v], force_ke[B,6,v]   (B=2, 6 faces, v=48^3)
static constexpr size_t OFF_ATOM = 0;
static constexpr size_t OFF_MMF  = (size_t)2*6*8*NVOX;
static constexpr size_t OFF_MKE  = OFF_MMF + (size_t)2*6*24*NVOX;
static constexpr size_t OFF_FMF  = OFF_MKE + (size_t)2*6*8*NVOX;
static constexpr size_t OFF_FKE  = OFF_FMF + (size_t)2*6*3*NVOX;

__device__ __forceinline__ v8f wmma16x16x32(v16h a, v16h b, v8f c) {
    return __builtin_amdgcn_wmma_f32_16x16x32_f16(false, a, false, b,
                                                  (short)0, c, false, false);
}

// fast silu: v * rcp(1+exp(-v))  -> v_exp_f32 + v_rcp_f32
__device__ __forceinline__ float silu_f(float v) {
    return v * __builtin_amdgcn_rcpf(1.f + __expf(-v));
}
// fast softplus: log(1+exp(v)) -> v_exp_f32 + v_log_f32 (guard overflow)
__device__ __forceinline__ float softplus_f(float v) {
    return (v > 20.f) ? v : __logf(1.f + __expf(v));
}

// 16 contiguous, 16B-aligned halves -> one WMMA fragment (2x ds_load_b128)
__device__ __forceinline__ v16h load_frag(const half_t* p) {
    v8h lo = *(const v8h*)p;
    v8h hh = *(const v8h*)(p + 8);
    return __builtin_shufflevector(lo, hh, 0,1,2,3,4,5,6,7,
                                           8,9,10,11,12,13,14,15);
}

__global__ __launch_bounds__(256)
void fused_face_mlp(const float* __restrict__ counts,
                    const float* __restrict__ momentum,
                    const float* __restrict__ ke,
                    const float* __restrict__ order,
                    const float* __restrict__ W1, const float* __restrict__ b1,
                    const float* __restrict__ W2, const float* __restrict__ b2,
                    const float* __restrict__ W3, const float* __restrict__ b3,
                    float* __restrict__ out)
{
    // Weights in *fragment order*: frag index -> 32 lanes x 16 halves contiguous
    __shared__ half_t sW1f[4*32*16];        // 4 N-tiles            (K=26 pad 32)
    __shared__ half_t sW2f[2*4*32*16];      // 2 K-chunks x 4 N-tiles
    __shared__ half_t sW3f[2*3*32*16];      // 2 K-chunks x 3 N-tiles (N=44 pad 48)
    __shared__ float  sB1[64], sB2[64], sB3[48];
    __shared__ half_t sX [NWAVES][16*32];   // input A-tile, row-major (M=16,K=32)
    __shared__ half_t sHf[NWAVES][2*32*16]; // hidden tile in A-fragment order

    const int tid  = threadIdx.x;
    const int lane = tid & 31;
    const int w    = tid >> 5;
    const int mrow = lane & 15;
    const int hi   = (lane >> 4) & 1;

    // ---- stage weights, pre-swizzled into B-fragment lane order ----
    // B 32x16 f16 layout: lanes 0-15 hold K=j (j<8) / 16+j-8, lanes 16-31 hold K=8+j (j<8) / 24+j-8
    for (int i = tid; i < 4*32*16; i += 256) {
        int j = i & 15, l = (i >> 4) & 31, nt = i >> 9;
        int K = (j & 7) + ((j >= 8) ? 16 : 0) + ((l >= 16) ? 8 : 0);
        int N = nt*16 + (l & 15);
        sW1f[i] = (K < 26) ? (half_t)W1[N*26 + K] : (half_t)0.f;
    }
    for (int i = tid; i < 2*4*32*16; i += 256) {
        int j = i & 15, l = (i >> 4) & 31, nt = (i >> 9) & 3, kc = i >> 11;
        int K = kc*32 + (j & 7) + ((j >= 8) ? 16 : 0) + ((l >= 16) ? 8 : 0);
        int N = nt*16 + (l & 15);
        sW2f[i] = (half_t)W2[N*64 + K];
    }
    for (int i = tid; i < 2*3*32*16; i += 256) {
        int j = i & 15, l = (i >> 4) & 31, r = i >> 9;
        int nt = r % 3, kc = r / 3;
        int K = kc*32 + (j & 7) + ((j >= 8) ? 16 : 0) + ((l >= 16) ? 8 : 0);
        int N = nt*16 + (l & 15);
        sW3f[i] = (N < 44) ? (half_t)W3[N*64 + K] : (half_t)0.f;
    }
    if (tid < 64) { sB1[tid] = b1[tid]; sB2[tid] = b2[tid]; }
    if (tid < 48) { sB3[tid] = (tid < 44) ? b3[tid] : 0.f; }
    __syncthreads();

    // ---- tile decomposition ----
    int t = blockIdx.x * NWAVES + w;      // < 2*6*48*48*3 = 82944 exactly
    const int zt   = t % 3;  t /= 3;
    const int y    = t % 48; t /= 48;
    const int x    = t % 48; t /= 48;
    const int face = t % 6;
    const int b    = t / 6;
    const int z0   = zt * 16;
    const int bfi  = b*6 + face;

    auto fetch = [&](int c, int sp) -> float {
        if (c < 8)   return counts  [(size_t)(b*8 + c)       * NVOX + sp];
        if (c < 11)  return momentum[(size_t)(b*3 + (c - 8)) * NVOX + sp];
        if (c == 11) return ke      [(size_t)b * NVOX + sp];
        return order[(size_t)b * NVOX + sp];
    };
    auto chanval = [&](int cc, int z) -> float {
        const bool dst = cc >= 13;
        const int  c   = dst ? cc - 13 : cc;
        int xx = x, yy = y, zz = z;
        if (dst) {
            if      (face == 0) xx = (x + 1)  % 48;
            else if (face == 1) xx = (x + 47) % 48;
            else if (face == 2) yy = (y + 1)  % 48;
            else if (face == 3) yy = (y + 47) % 48;
            else if (face == 4) { zz = z + 1; if (zz >= 48) return 0.f; }
            else                { zz = z - 1; if (zz <  0)  return 0.f; }
        }
        const int sp = xx*2304 + yy*48 + zz;
        if (c >= 8 && c < 11) {
            const int d  = c - 8;
            const int pm = (d == 0) ? (face >> 1)
                         : (d == 1) ? ((face < 2) ? 1 : 0)
                                    : ((face < 4) ? 2 : 1);
            const float sg = (d == 0 && (face & 1)) ? -1.f : 1.f;
            return sg * fetch(8 + pm, sp);
        }
        return fetch(c, sp);
    };

    // ---- gather input tile (coalesced along z) ----
    for (int i = lane; i < 16*32; i += 32) {
        int cc = i >> 4, m = i & 15;
        sX[w][m*32 + cc] = (cc < 26) ? (half_t)chanval(cc, z0 + m) : (half_t)0.f;
    }
    __syncthreads();

    // C(row,col) value -> sHf fragment slot, f16-converted with activation fused
    // A-frag mapping: k in chunk -> hi2=(k>>3)&1, j=((k>>1)&8)|(k&7)
    auto stash_h = [&](int nt, const v8f& c, const float* bias) {
        const int kc   = nt >> 1;
        const int kpos = (nt & 1)*16 + mrow;
        const int hi2  = (kpos >> 3) & 1;
        const int j    = ((kpos >> 1) & 8) | (kpos & 7);
        const float bb = bias[nt*16 + mrow];
        #pragma unroll
        for (int i = 0; i < 8; ++i) {
            int row = i + hi*8;
            sHf[w][(kc*32 + hi2*16 + row)*16 + j] = (half_t)silu_f(c[i] + bb);
        }
    };

    // ---- GEMM1: [16x32] x [32x64] ----
    const v16h a1 = [&]{
        v8h lo = *(const v8h*)&sX[w][mrow*32 + hi*8];
        v8h hh = *(const v8h*)&sX[w][mrow*32 + 16 + hi*8];
        return __builtin_shufflevector(lo, hh, 0,1,2,3,4,5,6,7,
                                               8,9,10,11,12,13,14,15);
    }();
    #pragma unroll
    for (int nt = 0; nt < 4; ++nt) {
        v16h bf = load_frag(&sW1f[(nt*32 + lane)*16]);
        v8f c = {};
        c = wmma16x16x32(a1, bf, c);
        stash_h(nt, c, sB1);
    }
    __syncthreads();

    // ---- GEMM2: [16x64] x [64x64] ----
    const v16h a2lo = load_frag(&sHf[w][(0*32 + lane)*16]);
    const v16h a2hi = load_frag(&sHf[w][(1*32 + lane)*16]);
    __syncthreads();                          // reads done before overwrite
    #pragma unroll
    for (int nt = 0; nt < 4; ++nt) {
        v16h blo = load_frag(&sW2f[((0*4 + nt)*32 + lane)*16]);
        v16h bhi = load_frag(&sW2f[((1*4 + nt)*32 + lane)*16]);
        v8f c = {};
        c = wmma16x16x32(a2lo, blo, c);
        c = wmma16x16x32(a2hi, bhi, c);
        stash_h(nt, c, sB2);
    }
    __syncthreads();

    // ---- GEMM3: [16x64] x [64x48], fused bias + activation + store ----
    const v16h a3lo = load_frag(&sHf[w][(0*32 + lane)*16]);
    const v16h a3hi = load_frag(&sHf[w][(1*32 + lane)*16]);
    const size_t sp_base = (size_t)(x*2304 + y*48 + z0 + hi*8);
    #pragma unroll
    for (int nt = 0; nt < 3; ++nt) {
        v16h blo = load_frag(&sW3f[((0*3 + nt)*32 + lane)*16]);
        v16h bhi = load_frag(&sW3f[((1*3 + nt)*32 + lane)*16]);
        v8f c = {};
        c = wmma16x16x32(a3lo, blo, c);
        c = wmma16x16x32(a3hi, bhi, c);

        const int o = nt*16 + mrow;           // output channel for this lane
        if (o < 44) {
            size_t plane; bool act;
            if      (o < 3)  { plane = OFF_FMF  + (size_t)(bfi*3  + o)        * NVOX; act = false; }
            else if (o == 3) { plane = OFF_FKE  + (size_t) bfi                 * NVOX; act = true;  }
            else if (o < 12) { plane = OFF_ATOM + (size_t)(bfi*8  + (o - 4))  * NVOX; act = true;  }
            else if (o < 36) { plane = OFF_MMF  + (size_t)(bfi*24 + (o - 12)) * NVOX; act = false; }
            else             { plane = OFF_MKE  + (size_t)(bfi*8  + (o - 36)) * NVOX; act = true;  }
            const float bias = sB3[o];
            float r[8];
            #pragma unroll
            for (int i = 0; i < 8; ++i) {
                float v = c[i] + bias;
                r[i] = act ? softplus_f(v) : v;
            }
            float4* p = (float4*)(out + plane + sp_base);   // 16B aligned
            p[0] = make_float4(r[0], r[1], r[2], r[3]);
            p[1] = make_float4(r[4], r[5], r[6], r[7]);
        }
    }
}

extern "C" void kernel_launch(void* const* d_in, const int* in_sizes, int n_in,
                              void* d_out, int out_size, void* d_ws, size_t ws_size,
                              hipStream_t stream) {
    (void)in_sizes; (void)n_in; (void)out_size; (void)d_ws; (void)ws_size;
    const float* counts   = (const float*)d_in[0];
    const float* momentum = (const float*)d_in[1];
    const float* kein     = (const float*)d_in[2];
    const float* orderin  = (const float*)d_in[3];
    const float* W1 = (const float*)d_in[4];
    const float* b1 = (const float*)d_in[5];
    const float* W2 = (const float*)d_in[6];
    const float* b2 = (const float*)d_in[7];
    const float* W3 = (const float*)d_in[8];
    const float* b3 = (const float*)d_in[9];

    const int tiles = 2 * 6 * 48 * 48 * 3;     // 82,944 M=16 tiles
    dim3 grid(tiles / NWAVES);                 // 10,368 blocks
    fused_face_mlp<<<grid, 256, 0, stream>>>(counts, momentum, kein, orderin,
                                             W1, b1, W2, b2, W3, b3,
                                             (float*)d_out);
}
